// FFTAlexNet_83107617177842
// MI455X (gfx1250) — compile-verified, matrix operand order checked by reading, not run on hardware
//
#include <hip/hip_runtime.h>

typedef unsigned short u16;
typedef unsigned int   u32;
typedef unsigned long long u64;

typedef __attribute__((ext_vector_type(16))) __bf16 bf16x16;
typedef __attribute__((ext_vector_type(8)))  __bf16 bf16x8;
typedef __attribute__((ext_vector_type(8)))  float  f32x8;

#define SHUF16(lo, hi) __builtin_shufflevector((lo), (hi), 0,1,2,3,4,5,6,7,8,9,10,11,12,13,14,15)
#define LDT 40  // padded LDS row stride in bf16 elements (80B: 16B aligned, bank-staggered)

__device__ __forceinline__ u16 f32_to_bf16(float f) {
  u32 u = __float_as_uint(f);
  u32 r = u + 0x7fffu + ((u >> 16) & 1u);   // round-to-nearest-even
  return (u16)(r >> 16);
}
__device__ __forceinline__ float bf16_to_f32(u16 h) {
  return __uint_as_float(((u32)h) << 16);
}

// CDNA5 async global->LDS copy (ASYNCcnt-tracked); generic LDS pointer's low
// 32 bits are the wave-relative LDS address expected in VDST.
__device__ __forceinline__ void async_load_b128(const void* g, void* lds) {
  u32 l = (u32)(uintptr_t)lds;
  u64 a = (u64)(uintptr_t)g;
  asm volatile("global_load_async_to_lds_b128 %0, %1, off" :: "v"(l), "v"(a) : "memory");
}
// Wait until at most N async ops remain in flight (per-wave, in-order complete).
template <int N>
__device__ __forceinline__ void wait_async_le() {
  asm volatile("s_wait_asynccnt %0" :: "i"(N) : "memory");
}

__global__ void cast_f32_bf16(const float* __restrict__ in, u16* __restrict__ out, int n) {
  int i = blockIdx.x * blockDim.x + threadIdx.x;
  if (i < n) out[i] = f32_to_bf16(in[i]);
}

// ---------------------------------------------------------------------------
// Implicit-GEMM conv. M = Cout (mult 64), N = B*H*W (mult 128), K = Cin*k*k.
// Block: 256 thr = 8 waves; tile 64x128x32; wave -> 2x2 accumulators (16x16).
// Double-buffered LDS pipeline: async DMA for step ks+1 overlaps WMMAs of ks.
// ---------------------------------------------------------------------------
__global__ __launch_bounds__(256)
void conv_wmma(const u16* __restrict__ X, const u16* __restrict__ Wt,
               const float* __restrict__ bias, u16* __restrict__ Y,
               int Cin, int Cout, int H, int W, int ksz, int pad)
{
  __shared__ u16 Alds[2][64 * LDT];
  __shared__ u16 Blds[2][128 * LDT];

  const int K   = Cin * ksz * ksz;
  const int HW  = H * W;
  const int tid = threadIdx.x;
  const int lane = tid & 31;
  const int wave = tid >> 5;
  const int wm = wave >> 2;      // 0..1 -> 32-row M sub-tile
  const int wn = wave & 3;       // 0..3 -> 32-col N sub-tile

  const int pixBase = blockIdx.x * 128;
  const int coBase  = blockIdx.y * 64;

  // A loader: 64 rows x 32 K, 8 elems (16B) per thread
  const int alr = tid >> 2, alk = (tid & 3) * 8;
  const int aRow = coBase + alr;
  const bool kAligned = (K & 31) == 0;   // conv2..5: async DMA fast path

  // B loader: 128 cols x 32 K, 16 elems per thread (im2col gather)
  const int bln = tid >> 1, blk = (tid & 1) * 16;
  const int p   = pixBase + bln;
  const int pb  = p / HW;
  const int phw = p - pb * HW;
  const int py  = phw / W;
  const int px  = phw - py * W;
  const int kk2 = ksz * ksz;

  f32x8 acc[2][2] = {};
  const int ksteps = (K + 31) >> 5;

  auto load_tiles = [&](int ks, int buf) {
    const int k0 = ks << 5;
    u16* A  = &Alds[buf][0];
    u16* Bt = &Blds[buf][0];
    // A tile (weights, contiguous rows)
    const long base = (long)aRow * K + k0 + alk;
    if (kAligned) {
      async_load_b128(&Wt[base], &A[alr * LDT + alk]);
    } else {
      __builtin_prefetch(&Wt[base + 32], 0, 0);   // global_prefetch_b8
      #pragma unroll
      for (int j = 0; j < 8; ++j)
        A[alr * LDT + alk + j] = (k0 + alk + j < K) ? Wt[base + j] : (u16)0;
    }
    // B tile (im2col gather), n-major transposed: Bt[n][kk]
    #pragma unroll
    for (int j = 0; j < 16; ++j) {
      const int kk = k0 + blk + j;
      u16 v = 0;
      if (kk < K) {
        const int ci  = kk / kk2;
        const int rem = kk - ci * kk2;
        const int ky  = rem / ksz;
        const int kx  = rem - ky * ksz;
        const int iy  = py + ky - pad;
        const int ix  = px + kx - pad;
        if ((unsigned)iy < (unsigned)H && (unsigned)ix < (unsigned)W)
          v = X[(((long)pb * Cin + ci) * H + iy) * W + ix];
      }
      Bt[bln * LDT + blk + j] = v;
    }
  };

  load_tiles(0, 0);
  for (int ks = 0; ks < ksteps; ++ks) {
    const int cur = ks & 1;
    const bool havenext = (ks + 1 < ksteps);
    if (havenext) load_tiles(ks + 1, cur ^ 1);   // DMA overlaps compute below
    if (kAligned) {
      if (havenext) wait_async_le<1>();          // current A-tile DMA done; next stays in flight
      else          wait_async_le<0>();
    }
    __syncthreads();

    // fragments (lanes0-15: K0-7/16-23, lanes16-31: K8-15/24-31)
    const int kb = (lane >> 4) * 8;
    const int ar0 = wm * 32 + (lane & 15);
    const int br0 = wn * 32 + (lane & 15);
    const u16* A  = &Alds[cur][0];
    const u16* Bt = &Blds[cur][0];
    bf16x16 af[2], bfr[2];
    af[0] = SHUF16(*(const bf16x8*)&A[ar0 * LDT + kb],
                   *(const bf16x8*)&A[ar0 * LDT + kb + 16]);
    af[1] = SHUF16(*(const bf16x8*)&A[(ar0 + 16) * LDT + kb],
                   *(const bf16x8*)&A[(ar0 + 16) * LDT + kb + 16]);
    bfr[0] = SHUF16(*(const bf16x8*)&Bt[br0 * LDT + kb],
                    *(const bf16x8*)&Bt[br0 * LDT + kb + 16]);
    bfr[1] = SHUF16(*(const bf16x8*)&Bt[(br0 + 16) * LDT + kb],
                    *(const bf16x8*)&Bt[(br0 + 16) * LDT + kb + 16]);
    #pragma unroll
    for (int mi = 0; mi < 2; ++mi)
      #pragma unroll
      for (int ni = 0; ni < 2; ++ni)
        acc[mi][ni] = __builtin_amdgcn_wmma_f32_16x16x32_bf16(
            false, af[mi], false, bfr[ni], (short)0, acc[mi][ni], false, false);
    __syncthreads();   // readers done before buffer `cur` is refilled
  }

  // epilogue: bias + ReLU, bf16 out in [B][Cout][H][W]
  const int ncol  = lane & 15;
  const int mhalf = (lane >> 4) * 8;   // C/D layout: VGPR r -> M=r (+8 hi half)
  #pragma unroll
  for (int mi = 0; mi < 2; ++mi) {
    #pragma unroll
    for (int r = 0; r < 8; ++r) {
      const int co = coBase + wm * 32 + mi * 16 + mhalf + r;
      const float bv = bias[co];
      #pragma unroll
      for (int ni = 0; ni < 2; ++ni) {
        const int pp = pixBase + wn * 32 + ni * 16 + ncol;
        const int b = pp / HW, hw = pp - b * HW;
        float v = acc[mi][ni][r] + bv; v = v > 0.f ? v : 0.f;
        Y[((long)b * Cout + co) * HW + hw] = f32_to_bf16(v);
      }
    }
  }
}

// ---------------------------------------------------------------------------
// FC GEMM: Y[n][m] = act( X[n][:] . Wt[m][:] + bias[m] ). K mult of 32.
// All tiles via async global->LDS b128, double-buffered. M guarded (fc3).
// ---------------------------------------------------------------------------
__global__ __launch_bounds__(256)
void fc_wmma(const u16* __restrict__ X, const u16* __restrict__ Wt,
             const float* __restrict__ bias, void* __restrict__ Yout,
             int M, int K, int relu, int out_f32)
{
  __shared__ u16 Alds[2][64 * LDT];
  __shared__ u16 Blds[2][128 * LDT];
  const int tid = threadIdx.x, lane = tid & 31, wave = tid >> 5;
  const int wm = wave >> 2, wn = wave & 3;
  const int nBase = blockIdx.x * 128;
  const int mBase = blockIdx.y * 64;
  const int alr = tid >> 2, alk = (tid & 3) * 8;
  int aRow = mBase + alr; if (aRow >= M) aRow = M - 1;  // clamped; dead rows discarded
  const int bln = tid >> 1, blk = (tid & 1) * 16;

  f32x8 acc[2][2] = {};
  const int ksteps = K >> 5;

  auto load_tiles = [&](int ks, int buf) {
    const int kg = ks << 5;
    const long abase = (long)aRow * K + kg + alk;
    const long bbase = (long)(nBase + bln) * K + kg + blk;
    async_load_b128(&Wt[abase],    &Alds[buf][alr * LDT + alk]);
    async_load_b128(&X[bbase],     &Blds[buf][bln * LDT + blk]);
    async_load_b128(&X[bbase + 8], &Blds[buf][bln * LDT + blk + 8]);
  };

  load_tiles(0, 0);
  for (int ks = 0; ks < ksteps; ++ks) {
    const int cur = ks & 1;
    const bool havenext = (ks + 1 < ksteps);
    if (havenext) load_tiles(ks + 1, cur ^ 1);   // 3 DMAs in flight over compute
    if (havenext) wait_async_le<3>();
    else          wait_async_le<0>();
    __syncthreads();

    const int kb = (lane >> 4) * 8;
    const int ar0 = wm * 32 + (lane & 15);
    const int br0 = wn * 32 + (lane & 15);
    const u16* A  = &Alds[cur][0];
    const u16* Bt = &Blds[cur][0];
    bf16x16 af[2], bfr[2];
    af[0] = SHUF16(*(const bf16x8*)&A[ar0 * LDT + kb],
                   *(const bf16x8*)&A[ar0 * LDT + kb + 16]);
    af[1] = SHUF16(*(const bf16x8*)&A[(ar0 + 16) * LDT + kb],
                   *(const bf16x8*)&A[(ar0 + 16) * LDT + kb + 16]);
    bfr[0] = SHUF16(*(const bf16x8*)&Bt[br0 * LDT + kb],
                    *(const bf16x8*)&Bt[br0 * LDT + kb + 16]);
    bfr[1] = SHUF16(*(const bf16x8*)&Bt[(br0 + 16) * LDT + kb],
                    *(const bf16x8*)&Bt[(br0 + 16) * LDT + kb + 16]);
    #pragma unroll
    for (int mi = 0; mi < 2; ++mi)
      #pragma unroll
      for (int ni = 0; ni < 2; ++ni)
        acc[mi][ni] = __builtin_amdgcn_wmma_f32_16x16x32_bf16(
            false, af[mi], false, bfr[ni], (short)0, acc[mi][ni], false, false);
    __syncthreads();
  }

  const int ncol  = lane & 15;
  const int mhalf = (lane >> 4) * 8;
  #pragma unroll
  for (int mi = 0; mi < 2; ++mi) {
    #pragma unroll
    for (int r = 0; r < 8; ++r) {
      const int m = mBase + wm * 32 + mi * 16 + mhalf + r;
      if (m >= M) continue;
      const float bv = bias[m];
      #pragma unroll
      for (int ni = 0; ni < 2; ++ni) {
        const int n = nBase + wn * 32 + ni * 16 + ncol;
        float v = acc[mi][ni][r] + bv;
        if (relu && v < 0.f) v = 0.f;
        if (out_f32) ((float*)Yout)[(long)n * M + m] = v;
        else         ((u16*) Yout)[(long)n * M + m] = f32_to_bf16(v);
      }
    }
  }
}

// 2x2 stride-2 maxpool on bf16 [B][C][H][W]
__global__ void maxpool2_bf16(const u16* __restrict__ in, u16* __restrict__ out,
                              int C, int H, int W, int total)
{
  int i = blockIdx.x * blockDim.x + threadIdx.x;
  if (i >= total) return;
  const int Wo = W >> 1, Ho = H >> 1;
  int x = i % Wo; int t = i / Wo;
  int y = t % Ho; t /= Ho;
  int c = t % C;  int b = t / C;
  const u16* base = in + (((long)b * C + c) * H + 2 * y) * W + 2 * x;
  float m = bf16_to_f32(base[0]);
  m = fmaxf(m, bf16_to_f32(base[1]));
  m = fmaxf(m, bf16_to_f32(base[W]));
  m = fmaxf(m, bf16_to_f32(base[W + 1]));
  out[i] = f32_to_bf16(m);
}

extern "C" void kernel_launch(void* const* d_in, const int* in_sizes, int n_in,
                              void* d_out, int out_size, void* d_ws, size_t ws_size,
                              hipStream_t stream)
{
  (void)in_sizes; (void)n_in; (void)out_size; (void)ws_size;
  const float* x    = (const float*)d_in[0];
  const float* w1   = (const float*)d_in[1];  const float* b1 = (const float*)d_in[2];
  const float* w2   = (const float*)d_in[3];  const float* b2 = (const float*)d_in[4];
  const float* w3   = (const float*)d_in[5];  const float* b3 = (const float*)d_in[6];
  const float* w4   = (const float*)d_in[7];  const float* b4 = (const float*)d_in[8];
  const float* w5   = (const float*)d_in[9];  const float* b5 = (const float*)d_in[10];
  const float* f1w  = (const float*)d_in[11]; const float* f1b = (const float*)d_in[12];
  const float* f2w  = (const float*)d_in[13]; const float* f2b = (const float*)d_in[14];
  const float* f3w  = (const float*)d_in[15]; const float* f3b = (const float*)d_in[16];

  size_t off = 0;
  auto alloc = [&](size_t elems) {
    u16* p = (u16*)d_ws + off;
    off += (elems + 127) & ~(size_t)127;
    return p;
  };
  u16* xbf  = alloc(786432);          // 256*3*32*32
  u16* wb1  = alloc(23232);           // 64*363
  u16* wb2  = alloc(602112);          // 192*3136
  u16* wb3  = alloc(1843200);         // 384*4800
  u16* wb4  = alloc(884736);          // 256*3456
  u16* wb5  = alloc(589824);          // 256*2304
  u16* fw1  = alloc(16777216);        // 4096*4096
  u16* fw2  = alloc(16777216);
  u16* fw3  = alloc(409600);          // 100*4096
  u16* buf0 = alloc(16777216);        // ping
  u16* buf1 = alloc(16777216);        // pong

  auto cast = [&](const float* in, u16* outp, int n) {
    cast_f32_bf16<<<(n + 255) / 256, 256, 0, stream>>>(in, outp, n);
  };
  cast(x,   xbf, 786432);
  cast(w1,  wb1, 23232);
  cast(w2,  wb2, 602112);
  cast(w3,  wb3, 1843200);
  cast(w4,  wb4, 884736);
  cast(w5,  wb5, 589824);
  cast(f1w, fw1, 16777216);
  cast(f2w, fw2, 16777216);
  cast(f3w, fw3, 409600);

  // conv1: 3->64, 32x32, k=11 p=5 ; N=262144
  conv_wmma<<<dim3(2048, 1), 256, 0, stream>>>(xbf, wb1, b1, buf0, 3, 64, 32, 32, 11, 5);
  maxpool2_bf16<<<(4194304 + 255) / 256, 256, 0, stream>>>(buf0, buf1, 64, 32, 32, 4194304);
  // conv2: 64->192, 16x16, k=7 p=3 ; N=65536
  conv_wmma<<<dim3(512, 3), 256, 0, stream>>>(buf1, wb2, b2, buf0, 64, 192, 16, 16, 7, 3);
  maxpool2_bf16<<<(3145728 + 255) / 256, 256, 0, stream>>>(buf0, buf1, 192, 16, 16, 3145728);
  // conv3: 192->384, 8x8, k=5 p=2 ; N=16384
  conv_wmma<<<dim3(128, 6), 256, 0, stream>>>(buf1, wb3, b3, buf0, 192, 384, 8, 8, 5, 2);
  // conv4: 384->256, 8x8, k=3 p=1
  conv_wmma<<<dim3(128, 4), 256, 0, stream>>>(buf0, wb4, b4, buf1, 384, 256, 8, 8, 3, 1);
  // conv5: 256->256, 8x8, k=3 p=1
  conv_wmma<<<dim3(128, 4), 256, 0, stream>>>(buf1, wb5, b5, buf0, 256, 256, 8, 8, 3, 1);
  maxpool2_bf16<<<(1048576 + 255) / 256, 256, 0, stream>>>(buf0, buf1, 256, 8, 8, 1048576);

  // FC stack: [256][4096] bf16 activations
  fc_wmma<<<dim3(2, 64), 256, 0, stream>>>(buf1, fw1, f1b, buf0, 4096, 4096, 1, 0);
  fc_wmma<<<dim3(2, 64), 256, 0, stream>>>(buf0, fw2, f2b, buf1, 4096, 4096, 1, 0);
  fc_wmma<<<dim3(2, 2),  256, 0, stream>>>(buf1, fw3, f3b, d_out, 100, 4096, 0, 1);
}
